// SwinTransformerStage_15874199126206
// MI455X (gfx1250) — matrix operand order repaired and checked by reading, not compiled, not run on hardware
//
#include <hip/hip_runtime.h>
#include <math.h>
#include <stdint.h>

// Problem constants (SwinV2 stage: B=32, H=W=56, C=128, WS=7, NH=4, DEPTH=2)
#define NTOK   100352   // 32*56*56 tokens
#define NWTOT  2048     // 32 * 64 windows
#define PM_M   25088    // 32*28*28 patch-merge rows

#define USE_ASYNC_LDS 1

typedef __attribute__((ext_vector_type(16))) __bf16       bf16x16;
typedef __attribute__((ext_vector_type(8)))  float        f32x8;
typedef __attribute__((ext_vector_type(4)))  float        f32x4;
typedef __attribute__((ext_vector_type(4)))  unsigned int u32x4;
typedef __attribute__((ext_vector_type(2)))  unsigned int u32x2;

union Frag { u32x4 q[2]; bf16x16 v; };

__device__ __forceinline__ unsigned short f32_to_bf16(float f){
  union { float f; unsigned int i; } x; x.f = f;
  unsigned int r = x.i + 0x7FFFu + ((x.i >> 16) & 1u);   // round-to-nearest-even
  return (unsigned short)(r >> 16);
}
__device__ __forceinline__ float bf16_to_f32(unsigned short u){
  union { unsigned int i; float f; } x; x.i = ((unsigned int)u) << 16;
  return x.f;
}
__device__ __forceinline__ unsigned int pack2(float a, float b){
  return (unsigned int)f32_to_bf16(a) | ((unsigned int)f32_to_bf16(b) << 16);
}
__device__ __forceinline__ f32x8 wmma_bf16(Frag a, Frag b, f32x8 c){
  return __builtin_amdgcn_wmma_f32_16x16x32_bf16(false, a.v, false, b.v, (short)0, c, false, false);
}

#if USE_ASYNC_LDS
// CDNA5 async global->LDS copy (per-lane 16B), tracked by ASYNCcnt.
__device__ __forceinline__ void async_b128(unsigned short* ldsDst, const unsigned short* gSrc){
  unsigned int       l = (unsigned int)(uintptr_t)ldsDst;       // LDS addr = addr[31:0]
  unsigned long long g = (unsigned long long)(uintptr_t)gSrc;
  asm volatile("global_load_async_to_lds_b128 %0, %1, off" :: "v"(l), "v"(g) : "memory");
}
__device__ __forceinline__ void async_wait0(){
  asm volatile("s_wait_asynccnt 0x0" ::: "memory");
}
#endif

// ---------------------------------------------------------------------------
// Generic bf16 WMMA GEMM: out[M,N] = act(A[M,K] @ Wt[N,K]^T + bias)
// 128 threads / 4 waves; block tile 64x128; wave tile 32x64 (2x4 WMMA accs);
// K templated so the K-loop fully unrolls.
// ---------------------------------------------------------------------------
template<int ACT, int OUTBF, int KDIM>
__global__ __launch_bounds__(128)
void gemm_wmma(const unsigned short* __restrict__ A,
               const unsigned short* __restrict__ Wt,
               const float* __restrict__ bias,
               float* __restrict__ outF,
               unsigned short* __restrict__ outB,
               int M, int N)
{
  __shared__ unsigned short As[64*32];
  __shared__ unsigned short Bs[128*32];
  const int tid   = threadIdx.x;
  const int lane  = tid & 31;
  const int wave  = tid >> 5;
  const int wm    = wave >> 1, wn = wave & 1;
  const int tileM = blockIdx.x * 64;
  const int tileN = blockIdx.y * 128;
  const int arow  = tid >> 1;
  const int ahalf = (tid & 1) * 16;
  const unsigned short* gAp = A  + (size_t)(tileM + arow) * KDIM + ahalf;
  const unsigned short* gBp = Wt + (size_t)(tileN + tid) * KDIM;

  f32x8 acc[2][4] = {};
#pragma unroll
  for (int k0 = 0; k0 < KDIM; k0 += 32){
#if USE_ASYNC_LDS
    __syncthreads();                       // prior iteration's LDS reads done
    async_b128(As + arow*32 + ahalf,     gAp + k0);
    async_b128(As + arow*32 + ahalf + 8, gAp + k0 + 8);
    async_b128(Bs + tid*32,              gBp + k0);
    async_b128(Bs + tid*32 + 8,          gBp + k0 + 8);
    async_b128(Bs + tid*32 + 16,         gBp + k0 + 16);
    async_b128(Bs + tid*32 + 24,         gBp + k0 + 24);
    async_wait0();
#else
    u32x4 a0 = ((const u32x4*)(gAp + k0))[0];
    u32x4 a1 = ((const u32x4*)(gAp + k0))[1];
    u32x4 b0 = ((const u32x4*)(gBp + k0))[0];
    u32x4 b1 = ((const u32x4*)(gBp + k0))[1];
    u32x4 b2 = ((const u32x4*)(gBp + k0))[2];
    u32x4 b3 = ((const u32x4*)(gBp + k0))[3];
    __syncthreads();
    *(u32x4*)(As + arow*32 + ahalf)     = a0;
    *(u32x4*)(As + arow*32 + ahalf + 8) = a1;
    *(u32x4*)(Bs + tid*32)      = b0;
    *(u32x4*)(Bs + tid*32 + 8)  = b1;
    *(u32x4*)(Bs + tid*32 + 16) = b2;
    *(u32x4*)(Bs + tid*32 + 24) = b3;
#endif
    __syncthreads();
    Frag af[2], bfr[4];
#pragma unroll
    for (int i=0;i<2;i++){
      int r  = wm*32 + i*16 + (lane & 15);
      int ko = (lane < 16) ? 0 : 8;     // A layout: K 0-7/16-23 vs 8-15/24-31
      af[i].q[0] = *(const u32x4*)(As + r*32 + ko);
      af[i].q[1] = *(const u32x4*)(As + r*32 + ko + 16);
    }
#pragma unroll
    for (int j=0;j<4;j++){
      int r  = wn*64 + j*16 + (lane & 15);
      int ko = (lane < 16) ? 0 : 8;     // B layout: K 0-7/16-23 vs 8-15/24-31
      bfr[j].q[0] = *(const u32x4*)(Bs + r*32 + ko);
      bfr[j].q[1] = *(const u32x4*)(Bs + r*32 + ko + 16);
    }
#pragma unroll
    for (int i=0;i<2;i++)
#pragma unroll
      for (int j=0;j<4;j++)
        acc[i][j] = wmma_bf16(af[i], bfr[j], acc[i][j]);
  }
#pragma unroll
  for (int i=0;i<2;i++){
#pragma unroll
    for (int j=0;j<4;j++){
      int mbase = tileM + wm*32 + i*16 + ((lane<16)?0:8);
      int ncol  = tileN + wn*64 + j*16 + (lane & 15);
      float bv  = bias ? bias[ncol] : 0.0f;
#pragma unroll
      for (int e=0;e<8;e++){
        float v = acc[i][j][e] + bv;
        if (ACT == 1) v = 0.5f * v * (1.0f + erff(v * 0.70710678118f)); // exact GELU
        size_t off = (size_t)(mbase + e) * N + ncol;
        if (OUTBF) outB[off] = f32_to_bf16(v);
        else       outF[off] = v;
      }
    }
  }
}

// ---------------------------------------------------------------------------
// Window attention: one wave per (window, head). 49 padded to 64.
// S = qn @ kn^T (16 WMMA), softmax in LDS (+bias, contiguous), out = P @ V.
// ---------------------------------------------------------------------------
__global__ __launch_bounds__(32)
void attn_wmma(const unsigned short* __restrict__ qn,   // [2048][4][64][32] bf16
               const unsigned short* __restrict__ kn,   // same
               const unsigned short* __restrict__ vt,   // [2048][4][32][64] bf16 (v^T)
               const float* __restrict__ btab,          // [nWb][4][64][64] combined bias
               const float* __restrict__ scales,        // [4]
               unsigned short* __restrict__ out,        // [2048*49,128] bf16 window-order
               int maskPerWin)
{
  __shared__ float          Sm[64*64];
  __shared__ unsigned short Pm[64*64];
  const int wh  = blockIdx.x;
  const int win = wh >> 2, h = wh & 3;
  const int lane = threadIdx.x;
  const unsigned short* q = qn + (size_t)wh * 2048;
  const unsigned short* k = kn + (size_t)wh * 2048;
  const unsigned short* v = vt + (size_t)wh * 2048;
  const int wmask = maskPerWin ? (win & 63) : 0;
  const float* bt = btab + ((size_t)wmask * 4 + h) * 4096;
  const float scale = scales[h];

  Frag qa[4], kb[4];
#pragma unroll
  for (int i=0;i<4;i++){
    int r   = i*16 + (lane & 15);
    int koA = (lane < 16) ? 0 : 8;
    qa[i].q[0] = *(const u32x4*)(q + r*32 + koA);
    qa[i].q[1] = *(const u32x4*)(q + r*32 + koA + 16);
    int koB = (lane < 16) ? 0 : 8;
    kb[i].q[0] = *(const u32x4*)(k + r*32 + koB);
    kb[i].q[1] = *(const u32x4*)(k + r*32 + koB + 16);
  }
#pragma unroll
  for (int i=0;i<4;i++){
#pragma unroll
    for (int j=0;j<4;j++){
      f32x8 s = {};
      s = wmma_bf16(qa[i], kb[j], s);
#pragma unroll
      for (int e=0;e<8;e++){
        int m = i*16 + ((lane<16) ? e : e+8);
        int n = j*16 + (lane & 15);
        Sm[m*64+n] = s[e]*scale;
      }
    }
  }
  __syncthreads();
  // softmax over key dim; bias added here with contiguous per-row reads
  // (padded cols carry -1e30 bias -> exp==0)
#pragma unroll
  for (int rr=0;rr<2;rr++){
    int m = lane*2 + rr;
    const float* bR = bt + m*64;
    const float* sR = Sm + m*64;
    float mx = -1e30f;
    for (int c2=0;c2<64;c2++) mx = fmaxf(mx, sR[c2] + bR[c2]);
    float sum = 0.f;
    for (int c2=0;c2<64;c2++) sum += expf(sR[c2] + bR[c2] - mx);
    float inv = 1.0f / sum;
    for (int c2=0;c2<64;c2++) Pm[m*64+c2] = f32_to_bf16(expf(sR[c2] + bR[c2] - mx)*inv);
  }
  __syncthreads();

  f32x8 O[4][2] = {};
#pragma unroll
  for (int kk=0; kk<2; kk++){
    Frag pa[4], vb[2];
#pragma unroll
    for (int i=0;i<4;i++){
      int r  = i*16 + (lane & 15);
      int ko = kk*32 + ((lane<16)?0:8);
      pa[i].q[0] = *(const u32x4*)(Pm + r*64 + ko);
      pa[i].q[1] = *(const u32x4*)(Pm + r*64 + ko + 16);
    }
#pragma unroll
    for (int j=0;j<2;j++){
      int col = j*16 + (lane & 15);
      int ko  = kk*32 + ((lane<16)?0:8);
      vb[j].q[0] = *(const u32x4*)(v + col*64 + ko);
      vb[j].q[1] = *(const u32x4*)(v + col*64 + ko + 16);
    }
#pragma unroll
    for (int i=0;i<4;i++)
#pragma unroll
      for (int j=0;j<2;j++)
        O[i][j] = wmma_bf16(pa[i], vb[j], O[i][j]);
  }
#pragma unroll
  for (int i=0;i<4;i++){
#pragma unroll
    for (int j=0;j<2;j++){
#pragma unroll
      for (int e=0;e<8;e++){
        int nq = i*16 + ((lane<16)? e : e+8);
        if (nq < 49){
          size_t off = ((size_t)win*49 + nq)*128 + h*32 + j*16 + (lane & 15);
          out[off] = f32_to_bf16(O[i][j][e]);
        }
      }
    }
  }
}

// ---------------------------------------------------------------------------
// Support kernels
// ---------------------------------------------------------------------------
__global__ void transpose_cvt(const float* __restrict__ in, unsigned short* __restrict__ out,
                              int K, int N){
  int idx = blockIdx.x*256 + threadIdx.x;
  if (idx >= K*N) return;
  int k = idx / N, n = idx % N;
  out[(size_t)n*K + k] = f32_to_bf16(in[idx]);
}

__global__ void build_qkv_bias(const float* __restrict__ qb, const float* __restrict__ vb,
                               float* __restrict__ out){
  int idx = blockIdx.x*256 + threadIdx.x;
  if (idx >= 768) return;
  int l = idx / 384, i = idx % 384;
  float v = 0.f;
  if (i < 128)       v = qb[l*128 + i];
  else if (i >= 256) v = vb[l*128 + i - 256];
  out[idx] = v;
}

__global__ void build_scales(const float* __restrict__ ls, float* __restrict__ out){
  int i = threadIdx.x;
  if (i < 8) out[i] = expf(fminf(ls[i], 4.6051702f));   // exp(min(logit_scale, ln 100))
}

__global__ void cpb_hidden_k(const float* __restrict__ w1, const float* __restrict__ b1,
                             float* __restrict__ hid){
  int i = blockIdx.x, j = threadIdx.x;
  float dy = ((float)(i/13) - 6.f) * (8.f/6.f);
  float dx = ((float)(i%13) - 6.f) * (8.f/6.f);
  float t0 = ((dy>0.f)?1.f:((dy<0.f)?-1.f:0.f)) * log2f(fabsf(dy)+1.f) * (1.f/3.f);
  float t1 = ((dx>0.f)?1.f:((dx<0.f)?-1.f:0.f)) * log2f(fabsf(dx)+1.f) * (1.f/3.f);
  float v = t0*w1[j] + t1*w1[512+j] + b1[j];
  hid[i*512+j] = v > 0.f ? v : 0.f;
}

__global__ void cpb_out_k(const float* __restrict__ hid, const float* __restrict__ w2,
                          float* __restrict__ val){
  int i = blockIdx.x, h = threadIdx.x;
  float s = 0.f;
  for (int j=0;j<512;j++) s += hid[i*512+j]*w2[j*4+h];
  val[i*4+h] = s;
}

__global__ void build_bias_tab(const float* __restrict__ cpbval, float* __restrict__ btab,
                               int shifted){
  int blk = blockIdx.x;
  int w = blk >> 2, h = blk & 3;
  float* dst = btab + (size_t)blk * 4096;
  for (int idx = threadIdx.x; idx < 4096; idx += 256){
    int n = idx >> 6, m = idx & 63;
    float v;
    if (m >= 49)      v = -1e30f;
    else if (n >= 49) v = 0.f;
    else {
      int i1=n/7, j1=n%7, i2=m/7, j2=m%7;
      int rel = (i1-i2+6)*13 + (j1-j2+6);
      float lv = cpbval[rel*4 + h];
      v = 16.f / (1.f + expf(-lv));
      if (shifted){
        int wh2 = w >> 3, ww2 = w & 7;
        int gy1 = wh2*7 + i1, gx1 = ww2*7 + j1;
        int gy2 = wh2*7 + i2, gx2 = ww2*7 + j2;
        int r1 = ((gy1<49)?0:((gy1<53)?1:2))*3 + ((gx1<49)?0:((gx1<53)?1:2));
        int r2 = ((gy2<49)?0:((gy2<53)?1:2))*3 + ((gx2<49)?0:((gx2<53)?1:2));
        if (r1 != r2) v += -100.f;
      }
    }
    dst[idx] = v;
  }
}

// roll(-shift) + window partition + f32->bf16, one wave per token
__global__ __launch_bounds__(256)
void partition_shift(const float* __restrict__ X, unsigned short* __restrict__ out, int shift){
  int token = blockIdx.x*8 + (threadIdx.x >> 5);
  int lane  = threadIdx.x & 31;
  int win = token / 49, tin = token % 49;
  int b = win >> 6, w = win & 63;
  int wh = w >> 3, ww2 = w & 7;
  int r = tin / 7, c = tin % 7;
  int ys = (wh*7 + r + shift) % 56;
  int xs = (ww2*7 + c + shift) % 56;
  f32x4 v = ((const f32x4*)(X + (((size_t)b*56 + ys)*56 + xs)*128))[lane];
  u32x2 pk; pk.x = pack2(v.x, v.y); pk.y = pack2(v.z, v.w);
  *(u32x2*)(out + (size_t)token*128 + lane*4) = pk;
}

__global__ __launch_bounds__(256)
void prep_qkv(const unsigned short* __restrict__ qkv,
              unsigned short* __restrict__ qn, unsigned short* __restrict__ kn,
              unsigned short* __restrict__ vt){
  int win = blockIdx.x;
  int t = threadIdx.x;
  if (t < 196){
    int h = t / 49, n = t % 49;
    size_t row = (size_t)win*49 + n;
    const unsigned short* qp = qkv + row*384 + h*32;
    const unsigned short* kp = qp + 128;
    const unsigned short* vp = qp + 256;
    float qv[32], kv[32], vv[32];
    float qs=0.f, ks=0.f;
    for (int d=0; d<32; d++){
      float qf = bf16_to_f32(qp[d]); qv[d]=qf; qs += qf*qf;
      float kf = bf16_to_f32(kp[d]); kv[d]=kf; ks += kf*kf;
      vv[d] = bf16_to_f32(vp[d]);
    }
    float qi = 1.f / fmaxf(sqrtf(qs), 1e-12f);
    float ki = 1.f / fmaxf(sqrtf(ks), 1e-12f);
    size_t base = ((size_t)win*4 + h) * 2048;
    for (int d=0; d<32; d++){
      qn[base + n*32 + d] = f32_to_bf16(qv[d]*qi);
      kn[base + n*32 + d] = f32_to_bf16(kv[d]*ki);
      vt[base + d*64 + n] = f32_to_bf16(vv[d]);
    }
  } else {
    int idx = t - 196;                       // zero-pad rows 49..63
    int h = idx / 15, n = 49 + idx % 15;
    size_t base = ((size_t)win*4 + h) * 2048;
    for (int d=0; d<32; d++){
      qn[base + n*32 + d] = 0;
      kn[base + n*32 + d] = 0;
      vt[base + d*64 + n] = 0;
    }
  }
}

// Post-norm residual: X += LN(y); optional window-reverse+unshift mapping of y,
// optional packed-bf16 copy of the updated X for the next GEMM.
__global__ __launch_bounds__(256)
void ln_residual(const float* __restrict__ yin, const float* __restrict__ g,
                 const float* __restrict__ bta, float* __restrict__ X,
                 unsigned short* __restrict__ Xb, int windowed, int shift){
  int token = blockIdx.x*8 + (threadIdx.x >> 5);
  int lane  = threadIdx.x & 31;
  size_t srow;
  if (windowed){
    int b = token / 3136, rem = token % 3136, y = rem/56, xg = rem%56;
    int ys = (y - shift + 56) % 56, xs = (xg - shift + 56) % 56;
    int wh = ys/7, r = ys%7, ww2 = xs/7, c = xs%7;
    srow = ((size_t)((b*8+wh)*8+ww2)*49 + r*7+c) * 128;
  } else srow = (size_t)token * 128;
  f32x4 v = ((const f32x4*)(yin + srow))[lane];
  float s = v.x+v.y+v.z+v.w;
  for (int o=16;o;o>>=1) s += __shfl_xor(s, o, 32);
  float mean = s * (1.f/128.f);
  f32x4 d = v - mean;
  float ss = d.x*d.x + d.y*d.y + d.z*d.z + d.w*d.w;
  for (int o=16;o;o>>=1) ss += __shfl_xor(ss, o, 32);
  float inv = rsqrtf(ss*(1.f/128.f) + 1e-5f);
  f32x4 g4 = ((const f32x4*)g)[lane];
  f32x4 b4 = ((const f32x4*)bta)[lane];
  size_t xo = (size_t)token*128 + lane*4;
  f32x4 xv = *(f32x4*)(X + xo);
  f32x4 o4 = xv + d*inv*g4 + b4;
  *(f32x4*)(X + xo) = o4;
  if (Xb){
    u32x2 pk; pk.x = pack2(o4.x, o4.y); pk.y = pack2(o4.z, o4.w);
    *(u32x2*)(Xb + xo) = pk;
  }
}

// PatchMerging gather (2x2 -> 4C) + LayerNorm(512) -> packed bf16 GEMM input
__global__ __launch_bounds__(256)
void patch_ln(const float* __restrict__ X, const float* __restrict__ g,
              const float* __restrict__ bt, unsigned short* __restrict__ out){
  int token = blockIdx.x*8 + (threadIdx.x >> 5);
  int lane  = threadIdx.x & 31;
  int b = token/784, rem = token%784, ho = rem/28, wo = rem%28;
  f32x4 vals[4];
  float s = 0.f;
#pragma unroll
  for (int u=0;u<4;u++){
    int qq = lane*16 + u*4;
    int i = (qq>>8)&1, j=(qq>>7)&1, c=qq&127;
    const float* src = X + (((size_t)(b*56 + ho*2+i))*56 + wo*2+j)*128 + c;
    vals[u] = *(const f32x4*)src;
    s += vals[u].x+vals[u].y+vals[u].z+vals[u].w;
  }
  for (int o=16;o;o>>=1) s += __shfl_xor(s, o, 32);
  float mean = s * (1.f/512.f);
  float ss = 0.f;
#pragma unroll
  for (int u=0;u<4;u++){
    f32x4 d = vals[u]-mean;
    ss += d.x*d.x + d.y*d.y + d.z*d.z + d.w*d.w;
  }
  for (int o=16;o;o>>=1) ss += __shfl_xor(ss, o, 32);
  float inv = rsqrtf(ss*(1.f/512.f) + 1e-5f);
#pragma unroll
  for (int u=0;u<4;u++){
    int qq = lane*16 + u*4;
    f32x4 g4 = *(const f32x4*)(g + qq);
    f32x4 b4 = *(const f32x4*)(bt + qq);
    f32x4 r  = (vals[u]-mean)*inv*g4 + b4;
    u32x2 pk; pk.x = pack2(r.x, r.y); pk.y = pack2(r.z, r.w);
    *(u32x2*)(out + (size_t)token*512 + qq) = pk;
  }
}

// ---------------------------------------------------------------------------
extern "C" void kernel_launch(void* const* d_in, const int* in_sizes, int n_in,
                              void* d_out, int out_size, void* d_ws, size_t ws_size,
                              hipStream_t stream)
{
  (void)in_sizes; (void)n_in; (void)out_size; (void)ws_size;
  const float* x_in   = (const float*)d_in[0];
  const float* qkv_w  = (const float*)d_in[1];
  const float* q_bias = (const float*)d_in[2];
  const float* v_bias = (const float*)d_in[3];
  const float* lscale = (const float*)d_in[4];
  const float* cpb_w1 = (const float*)d_in[5];
  const float* cpb_b1 = (const float*)d_in[6];
  const float* cpb_w2 = (const float*)d_in[7];
  const float* proj_w = (const float*)d_in[8];
  const float* proj_b = (const float*)d_in[9];
  const float* n1g = (const float*)d_in[10];
  const float* n1b = (const float*)d_in[11];
  const float* mw1 = (const float*)d_in[12];
  const float* mb1 = (const float*)d_in[13];
  const float* mw2 = (const float*)d_in[14];
  const float* mb2 = (const float*)d_in[15];
  const float* n2g = (const float*)d_in[16];
  const float* n2b = (const float*)d_in[17];
  const float* pmg = (const float*)d_in[18];
  const float* pmb = (const float*)d_in[19];
  const float* pmw = (const float*)d_in[20];

  char* p = (char*)d_ws;
  auto alloc = [&](size_t n)->char*{ char* r = p; p += (n + 255) & ~((size_t)255); return r; };

  float*          Xf    = (float*)alloc((size_t)NTOK*128*4);
  unsigned short* actB  = (unsigned short*)alloc((size_t)NTOK*128*2);
  unsigned short* qkvB  = (unsigned short*)alloc((size_t)NTOK*384*2);
  unsigned short* qnB   = (unsigned short*)alloc((size_t)NWTOT*4*64*32*2);
  unsigned short* knB   = (unsigned short*)alloc((size_t)NWTOT*4*64*32*2);
  unsigned short* vtB   = (unsigned short*)alloc((size_t)NWTOT*4*32*64*2);
  unsigned short* aoB   = (unsigned short*)alloc((size_t)NTOK*128*2);
  float*          tmpF  = (float*)alloc((size_t)NTOK*128*4);
  unsigned short* hidB  = (unsigned short*)alloc((size_t)NTOK*512*2);
  float*          btab  = (float*)alloc((size_t)64*4*64*64*4);
  float*          cpbh  = (float*)alloc((size_t)169*512*4);
  float*          cpbv  = (float*)alloc((size_t)169*4*4);
  float*          scal  = (float*)alloc(8*4);
  float*          qkvbi = (float*)alloc(768*4);
  unsigned short* pmin  = (unsigned short*)alloc((size_t)PM_M*512*2);
  unsigned short* qkvWt = (unsigned short*)alloc((size_t)2*384*128*2);
  unsigned short* projWt= (unsigned short*)alloc((size_t)2*128*128*2);
  unsigned short* w1t   = (unsigned short*)alloc((size_t)2*512*128*2);
  unsigned short* w2t   = (unsigned short*)alloc((size_t)2*128*512*2);
  unsigned short* pmWt  = (unsigned short*)alloc((size_t)256*512*2);

  hipMemcpyAsync(Xf, x_in, (size_t)NTOK*128*4, hipMemcpyDeviceToDevice, stream);

  for (int l=0;l<2;l++){
    transpose_cvt<<<(128*384+255)/256,256,0,stream>>>(qkv_w + (size_t)l*128*384, qkvWt + (size_t)l*384*128, 128, 384);
    transpose_cvt<<<(128*128+255)/256,256,0,stream>>>(proj_w + (size_t)l*128*128, projWt + (size_t)l*128*128, 128, 128);
    transpose_cvt<<<(128*512+255)/256,256,0,stream>>>(mw1 + (size_t)l*128*512, w1t + (size_t)l*512*128, 128, 512);
    transpose_cvt<<<(512*128+255)/256,256,0,stream>>>(mw2 + (size_t)l*512*128, w2t + (size_t)l*128*512, 512, 128);
  }
  transpose_cvt<<<(512*256+255)/256,256,0,stream>>>(pmw, pmWt, 512, 256);
  build_scales<<<1,8,0,stream>>>(lscale, scal);
  build_qkv_bias<<<3,256,0,stream>>>(q_bias, v_bias, qkvbi);

  for (int l=0;l<2;l++){
    int shift = (l & 1) ? 3 : 0;
    int nWb   = shift ? 64 : 1;
    cpb_hidden_k<<<169,512,0,stream>>>(cpb_w1 + (size_t)l*2*512, cpb_b1 + (size_t)l*512, cpbh);
    cpb_out_k<<<169,4,0,stream>>>(cpbh, cpb_w2 + (size_t)l*512*4, cpbv);
    build_bias_tab<<<nWb*4,256,0,stream>>>(cpbv, btab, shift ? 1 : 0);

    partition_shift<<<NTOK/8,256,0,stream>>>(Xf, actB, shift);
    gemm_wmma<0,1,128><<<dim3(NTOK/64,384/128),128,0,stream>>>(actB, qkvWt + (size_t)l*384*128,
        qkvbi + l*384, nullptr, qkvB, NTOK, 384);
    prep_qkv<<<NWTOT,256,0,stream>>>(qkvB, qnB, knB, vtB);
    attn_wmma<<<NWTOT*4,32,0,stream>>>(qnB, knB, vtB, btab, scal + l*4, aoB, shift ? 1 : 0);
    gemm_wmma<0,0,128><<<dim3(NTOK/64,128/128),128,0,stream>>>(aoB, projWt + (size_t)l*128*128,
        proj_b + l*128, tmpF, nullptr, NTOK, 128);
    ln_residual<<<NTOK/8,256,0,stream>>>(tmpF, n1g + l*128, n1b + l*128, Xf, actB, 1, shift);
    gemm_wmma<1,1,128><<<dim3(NTOK/64,512/128),128,0,stream>>>(actB, w1t + (size_t)l*512*128,
        mb1 + l*512, nullptr, hidB, NTOK, 512);
    gemm_wmma<0,0,512><<<dim3(NTOK/64,128/128),128,0,stream>>>(hidB, w2t + (size_t)l*128*512,
        mb2 + l*128, tmpF, nullptr, NTOK, 128);
    ln_residual<<<NTOK/8,256,0,stream>>>(tmpF, n2g + l*128, n2b + l*128, Xf, nullptr, 0, 0);
  }

  patch_ln<<<PM_M/8,256,0,stream>>>(Xf, pmg, pmb, pmin);
  gemm_wmma<0,0,512><<<dim3(PM_M/64,256/128),128,0,stream>>>(pmin, pmWt, nullptr,
      (float*)d_out, nullptr, PM_M, 256);
}
